// E34DiagonalWh_79551384257207
// MI455X (gfx1250) — matrix operand order repaired and checked by reading, not compiled, not run on hardware
//
#include <hip/hip_runtime.h>
#include <hip/hip_bf16.h>
#include <cstdint>
#include <cstddef>

// ---------------------------------------------------------------------------
// Shapes (fixed by the reference)
// ---------------------------------------------------------------------------
#define DIM_ 1024
#define DI_  2048   // D_INNER
#define B_   8
#define T_   2048
#define M_   (B_*T_)   // 16384 rows in all GEMMs

typedef __bf16 bf16_t;
typedef __attribute__((ext_vector_type(16))) __bf16 v16bf;
typedef __attribute__((ext_vector_type(8)))  __bf16 v8bf;
typedef __attribute__((ext_vector_type(8)))  float  v8f;
typedef unsigned int u32x4 __attribute__((ext_vector_type(4)));
typedef int          i32x8 __attribute__((ext_vector_type(8)));
typedef int          i32x4 __attribute__((ext_vector_type(4)));

#if defined(__has_builtin)
#  if __has_builtin(__builtin_amdgcn_tensor_load_to_lds)
#    define USE_TDM 1
#  endif
#endif
#ifndef USE_TDM
#  define USE_TDM 0
#endif

__device__ __forceinline__ bf16_t f2bf(float f) {
  union { float f; unsigned u; } in; in.f = f;
  unsigned r = (in.u + 0x7FFFu + ((in.u >> 16) & 1u)) >> 16;   // RNE
  union { unsigned short s; bf16_t b; } out; out.s = (unsigned short)r;
  return out.b;
}

__device__ __forceinline__ float sigm(float x) { return 1.0f / (1.0f + __expf(-x)); }

__device__ __forceinline__ float fast_tanh(float x) {
#if defined(__has_builtin) && __has_builtin(__builtin_amdgcn_tanhf)
  return __builtin_amdgcn_tanhf(x);           // V_TANH_F32 (gfx1250 TRANS op)
#else
  float y;
  asm volatile("v_tanh_f32 %0, %1\n\tv_nop" : "=v"(y) : "v"(x));  // +hazard slot
  return y;
#endif
}

// ---------------------------------------------------------------------------
// fp32 -> bf16 bulk convert (4 elems / thread)
// ---------------------------------------------------------------------------
__global__ __launch_bounds__(256) void f32_to_bf16_kernel(
    const float* __restrict__ in, bf16_t* __restrict__ out, int n) {
  int i = (blockIdx.x * blockDim.x + threadIdx.x) * 4;
  if (i + 3 < n) {
    float4 v = *(const float4*)(in + i);
    union { bf16_t b[4]; uint2 u; } pk;
    pk.b[0] = f2bf(v.x); pk.b[1] = f2bf(v.y);
    pk.b[2] = f2bf(v.z); pk.b[3] = f2bf(v.w);
    *(uint2*)(out + i) = pk.u;
  } else {
    for (; i < n; ++i) out[i] = f2bf(in[i]);
  }
}

// ---------------------------------------------------------------------------
// LDS tile geometry for the GEMM.
//   Block tile: 128 (M) x 256 (N), K-step 64.
//   A tile: 128 rows x 64 bf16  -> 128B data + 16B pad = 144B row stride
//   B tile: 256 rows x 64 bf16  -> same padded stride
//   Padding (16B per 128B) makes ds_load_b128 rows land in distinct banks.
// ---------------------------------------------------------------------------
#define KSTEP   64
#define ROWB    144
#define A_ROWS  128
#define B_ROWS  256
#define ASZ     (A_ROWS*ROWB)          // 18432 B
#define BSZ     (B_ROWS*ROWB)          // 36864 B
#define BUFSZ   (ASZ+BSZ)              // 55296 B (x2 buffers = 108 KiB LDS)

#if USE_TDM
// Issue one TDM 2D tile load: tile_rows x 64 bf16 elements from a row-major
// [rows, K] tensor into padded LDS rows. Descriptor per CDNA5 D# layout.
// This toolchain's builtin is the 6-arg form:
//   (u32x4 g0, i32x8 g1, i32x4 g2, i32x4 g3, i32x8 pad, i32 cpol)
__device__ __forceinline__ void tdm_load_tile(unsigned lds_off,
                                              const bf16_t* gptr,
                                              int tdim0, int stride0,
                                              int tile_rows) {
  unsigned long long ga = (unsigned long long)(uintptr_t)gptr;
  u32x4 g0;
  g0[0] = 1u;                                        // count=1 (valid user D#)
  g0[1] = lds_off;                                   // lds_addr (bytes)
  g0[2] = (unsigned)(ga & 0xFFFFFFFFu);              // global_addr[31:0]
  g0[3] = (unsigned)((ga >> 32) & 0x01FFFFFFu)       // global_addr[56:32]
        | (2u << 30);                                // type=2 ("image")
  i32x8 g1;
  g1[0] = (1 << 16)                                  // data_size=1 -> 2 bytes
        | (1 << 20)                                  // pad_enable
        | (4 << 22)                                  // pad_interval: 32 dwords
        | (3 << 25);                                 // pad_amount: 4 dwords
  g1[1] = (tdim0 & 0xFFFF) << 16;                    // tensor_dim0 [63:48]
  g1[2] = ((tdim0 >> 16) & 0xFFFF)                   // tensor_dim0 hi
        | ((tile_rows & 0xFFFF) << 16);              // tensor_dim1 lo
  g1[3] = ((tile_rows >> 16) & 0xFFFF)               // tensor_dim1 hi
        | (KSTEP << 16);                             // tile_dim0 = 64 elems
  g1[4] = tile_rows & 0xFFFF;                        // tile_dim1; tile_dim2=0
  g1[5] = stride0;                                   // tensor_dim0_stride lo32
  g1[6] = 0;                                         // stride0 hi; dim1_stride
  g1[7] = 0;
  i32x4 z4; z4[0] = 0; z4[1] = 0; z4[2] = 0; z4[3] = 0;
  i32x8 z8;
#pragma unroll
  for (int e = 0; e < 8; ++e) z8[e] = 0;
  __builtin_amdgcn_tensor_load_to_lds(g0, g1, z4, z4, z8, 0);
}
#else
// Fallback: cooperative global->LDS copy into the same padded layout.
__device__ __forceinline__ void coop_copy_tile(char* dst, const bf16_t* gsrc,
                                               int K, int rows, int tid) {
  const int chunks = rows * (KSTEP * 2 / 16);        // 16B chunks per row = 8
  for (int c = tid; c < chunks; c += 256) {
    const int row = c >> 3, cc = c & 7;
    *(uint4*)(dst + row * ROWB + cc * 16) =
        *(const uint4*)((const char*)(gsrc + (size_t)row * K) + cc * 16);
  }
}
#endif

// ---------------------------------------------------------------------------
// NT GEMM with WMMA bf16: C[m,n] = act( sum_k A[m,k] * Bm[n,k] )
//   A : [M,K] bf16 row-major, Bm : [N,K] bf16 row-major.
//   Block = 256 threads = 8 waves (2 M x 4 N); each wave owns a 64x64 tile
//   (4x4 fragments). K-tiles staged in double-buffered LDS by the TDM.
//   MODE 0: SiLU -> bf16 out   MODE 1: +bias[col] -> f32 out   MODE 2: f32 out
// ---------------------------------------------------------------------------
template <int MODE>
__global__ __launch_bounds__(256) void gemm_nt_wmma(
    const bf16_t* __restrict__ A,
    const bf16_t* __restrict__ Bm,
    const float*  __restrict__ bias,
    bf16_t* __restrict__ Cb,
    float*  __restrict__ Cf,
    int M, int N, int K) {
  __shared__ __align__(16) char smem[2 * BUFSZ];

  const int lane = threadIdx.x & 31;
  const int wave = threadIdx.x >> 5;
  const int wy   = wave >> 2;        // 0..1  (M within block tile)
  const int wx   = wave & 3;         // 0..3  (N within block tile)
  const int half = lane >> 4;        // 0: lanes 0-15, 1: lanes 16-31
  const int rl   = lane & 15;
  const int m0   = blockIdx.y * 128 + wy * 64;
  const int n0   = blockIdx.x * 256 + wx * 64;

  const bf16_t* Ag = A  + (size_t)blockIdx.y * 128 * K;   // block's A rows
  const bf16_t* Bg = Bm + (size_t)blockIdx.x * 256 * K;   // block's B rows
  const int nsteps = K / KSTEP;

  v8f acc[4][4];
#pragma unroll
  for (int i = 0; i < 4; ++i)
#pragma unroll
    for (int j = 0; j < 4; ++j)
#pragma unroll
      for (int e = 0; e < 8; ++e) acc[i][j][e] = 0.0f;

  // Per-lane LDS fragment base offsets (bytes).
  // A frag (16x32): lanes 0-15 hold M=rl, K=[0..7]+[16..23]; lanes 16-31
  //                 hold K=[8..15]+[24..31] -> half*16B, hi half at +32B.
  const int aBase = (wy * 64 + rl) * ROWB + half * 16;
  // B frag (32x16): lanes 0-15 hold N=rl, K=[0..7]+[16..23]; lanes 16-31
  //                 hold K=[8..15]+[24..31] -> half*16B, hi half at +32B.
  const int bBase = (wx * 64 + rl) * ROWB + half * 16;
  const unsigned ldsBase = (unsigned)(uintptr_t)&smem[0];

#if USE_TDM
  if (wave == 0) {   // prologue: fill buffer 0 (wave-level TDM ops)
    tdm_load_tile(ldsBase,       Ag, K, K, A_ROWS);
    tdm_load_tile(ldsBase + ASZ, Bg, K, K, B_ROWS);
  }
#endif

  for (int i = 0; i < nsteps; ++i) {
    const char* bufA = smem + (i & 1) * BUFSZ;
    const char* bufB = bufA + ASZ;

#if USE_TDM
    if (wave == 0) {
      if (i + 1 < nsteps) {       // issue next K-tile into the other buffer
        const unsigned nb = ldsBase + ((i + 1) & 1) * BUFSZ;
        const int kn = (i + 1) * KSTEP;
        tdm_load_tile(nb,       Ag + kn, K, K, A_ROWS);
        tdm_load_tile(nb + ASZ, Bg + kn, K, K, B_ROWS);
        __builtin_amdgcn_s_wait_tensorcnt(2);   // current tile's 2 loads done
      } else {
        __builtin_amdgcn_s_wait_tensorcnt(0);
      }
    }
    __syncthreads();              // current buffer visible to all waves
#else
    __syncthreads();              // previous iteration done reading LDS
    coop_copy_tile((char*)bufA, Ag + i * KSTEP, K, A_ROWS, (int)threadIdx.x);
    coop_copy_tile((char*)bufB, Bg + i * KSTEP, K, B_ROWS, (int)threadIdx.x);
    __syncthreads();
#endif

#pragma unroll
    for (int s = 0; s < 2; ++s) {           // two 16x16x32 sub-steps per tile
      v16bf af[4], bq[4];
#pragma unroll
      for (int ii = 0; ii < 4; ++ii) {
        const char* p = bufA + aBase + ii * (16 * ROWB) + s * 64;
        v8bf lo = *(const v8bf*)(p);
        v8bf hi = *(const v8bf*)(p + 32);
#pragma unroll
        for (int e = 0; e < 8; ++e) { af[ii][e] = lo[e]; af[ii][8 + e] = hi[e]; }
      }
#pragma unroll
      for (int jj = 0; jj < 4; ++jj) {
        const char* p = bufB + bBase + jj * (16 * ROWB) + s * 64;
        v8bf lo = *(const v8bf*)(p);
        v8bf hi = *(const v8bf*)(p + 32);
#pragma unroll
        for (int e = 0; e < 8; ++e) { bq[jj][e] = lo[e]; bq[jj][8 + e] = hi[e]; }
      }
#pragma unroll
      for (int ii = 0; ii < 4; ++ii)
#pragma unroll
        for (int jj = 0; jj < 4; ++jj)
          acc[ii][jj] = __builtin_amdgcn_wmma_f32_16x16x32_bf16(
              false, af[ii], false, bq[jj], (short)0, acc[ii][jj], false, false);
    }
#if USE_TDM
    __syncthreads();              // all waves done before TDM overwrites buffer
#endif
  }

  // C/D fragment layout: VGPR v holds M = v + half*8, N = rl
#pragma unroll
  for (int i = 0; i < 4; ++i) {
#pragma unroll
    for (int j = 0; j < 4; ++j) {
#pragma unroll
      for (int v = 0; v < 8; ++v) {
        const int row = m0 + i * 16 + half * 8 + v;
        const int col = n0 + j * 16 + rl;
        const float val = acc[i][j][v];
        if constexpr (MODE == 0) {
          Cb[(size_t)row * N + col] = f2bf(val * sigm(val));       // SiLU
        } else if constexpr (MODE == 1) {
          Cf[(size_t)row * N + col] = val + bias[col];
        } else {
          Cf[(size_t)row * N + col] = val;
        }
      }
    }
  }
}

// ---------------------------------------------------------------------------
// Diagonal recurrence: one thread per (b, feature) channel, sequential in T.
//   h_t = tanh(u_t + d_c*h_{t-1});  out_t = h_t * silu(h_t) = h^2 * sigmoid(h)
// Critical path per step = 1 FMA + 1 V_TANH_F32; gate/store off the chain.
// ---------------------------------------------------------------------------
__global__ __launch_bounds__(256) void scan_kernel(
    const float* __restrict__ U,     // [B,T,DI] fp32
    const float* __restrict__ h0,    // [B,DI]
    const float* __restrict__ dvec,  // [DI]
    bf16_t* __restrict__ outs,       // [B,T,DI] bf16 (GEMM3 input)
    float*  __restrict__ hfin) {     // [B,DI]
  const int idx = blockIdx.x * blockDim.x + threadIdx.x;  // b*DI + f
  const int f = idx & (DI_ - 1);
  const int b = idx >> 11;                                // DI_ == 2^11
  float dc = dvec[f];
  dc = fminf(0.99f, fmaxf(-0.99f, dc));
  float h = h0[idx];
  const float* up = U    + (size_t)b * T_ * DI_ + f;
  bf16_t*      op = outs + (size_t)b * T_ * DI_ + f;
  for (int t = 0; t < T_; ++t) {
    const float u = *up;
    h = fast_tanh(fmaf(dc, h, u));
    *op = f2bf(h * (h * sigm(h)));
    up += DI_; op += DI_;
  }
  hfin[idx] = h;
}

// ---------------------------------------------------------------------------
// Host-side orchestration
// ---------------------------------------------------------------------------
extern "C" void kernel_launch(void* const* d_in, const int* in_sizes, int n_in,
                              void* d_out, int out_size, void* d_ws, size_t ws_size,
                              hipStream_t stream) {
  (void)in_sizes; (void)n_in; (void)out_size; (void)ws_size;
  const float* x     = (const float*)d_in[0];
  const float* h0    = (const float*)d_in[1];
  const float* W_in  = (const float*)d_in[2];
  const float* W_x   = (const float*)d_in[3];
  const float* dvec  = (const float*)d_in[4];
  const float* bias  = (const float*)d_in[5];
  const float* W_out = (const float*)d_in[6];
  float* out = (float*)d_out;   // [B,T,DIM] then [B,DI] h_final

  // Workspace layout (bytes), ~304 MiB total:
  char* ws = (char*)d_ws;
  bf16_t* xbf    = (bf16_t*)(ws + 0);            // 16384*1024*2 = 32 MiB
  bf16_t* winbf  = (bf16_t*)(ws + 33554432);     // 2048*1024*2  =  4 MiB
  bf16_t* wxbf   = (bf16_t*)(ws + 37748736);     // 2048*2048*2  =  8 MiB
  bf16_t* woutbf = (bf16_t*)(ws + 46137344);     // 1024*2048*2  =  4 MiB
  bf16_t* xproj  = (bf16_t*)(ws + 50331648);     // 16384*2048*2 = 64 MiB
  float*  u      = (float*) (ws + 117440512);    // 16384*2048*4 = 128 MiB
  bf16_t* outsb  = (bf16_t*)(ws + 251658240);    // 16384*2048*2 = 64 MiB

  auto cvt = [&](const float* src, bf16_t* dst, int n) {
    int blocks = (n / 4 + 255) / 256;
    f32_to_bf16_kernel<<<blocks, 256, 0, stream>>>(src, dst, n);
  };
  cvt(x,     xbf,    M_  * DIM_);
  cvt(W_in,  winbf,  DI_ * DIM_);
  cvt(W_x,   wxbf,   DI_ * DI_);
  cvt(W_out, woutbf, DIM_ * DI_);

  // GEMM1: xproj = silu(x @ W_in^T)           [M, DI] bf16
  gemm_nt_wmma<0><<<dim3(DI_ / 256, M_ / 128), 256, 0, stream>>>(
      xbf, winbf, nullptr, xproj, nullptr, M_, DI_, DIM_);
  // GEMM2: u = xproj @ W_x^T + b              [M, DI] fp32
  gemm_nt_wmma<1><<<dim3(DI_ / 256, M_ / 128), 256, 0, stream>>>(
      xproj, wxbf, bias, nullptr, u, M_, DI_, DI_);
  // Recurrence -> outs (bf16) and h_final (fp32 tail of d_out)
  scan_kernel<<<(B_ * DI_) / 256, 256, 0, stream>>>(
      u, h0, dvec, outsb, out + (size_t)M_ * DIM_);
  // GEMM3: out = outs @ W_out^T               [M, DIM] fp32
  gemm_nt_wmma<2><<<dim3(DIM_ / 256, M_ / 128), 256, 0, stream>>>(
      outsb, woutbf, nullptr, nullptr, out, M_, DIM_, DI_);
}
